// Attention_1185410974455
// MI455X (gfx1250) — matrix operand order repaired and checked by reading, not compiled, not run on hardware
//
#include <hip/hip_runtime.h>
#include <hip/hip_bf16.h>

// ---------------------------------------------------------------------------
// CDNA5 (gfx1250) wave32 WMMA attention block, round 2.
//   GEMMs     : 8-wave workgroups, 128x64 tiles, TDM tensor_load_to_lds
//               double-buffered staging (s_wait_tensorcnt), WMMA from LDS.
//   RoPE      : elementwise; V is written TRANSPOSED so the P.V B-fragment
//               becomes a contiguous 32B load per lane.
//   Attention : flash-style online softmax, WMMA scores + WMMA P.V,
//               global_prefetch on the next key tile.
// ---------------------------------------------------------------------------

typedef __attribute__((ext_vector_type(8)))  _Float16 v8h;
typedef __attribute__((ext_vector_type(16))) _Float16 v16h;
typedef __attribute__((ext_vector_type(8)))  float    v8f;
typedef __attribute__((ext_vector_type(4)))  unsigned u32x4;
typedef __attribute__((ext_vector_type(8)))  int      i32x8;
typedef __attribute__((ext_vector_type(4)))  int      i32x4;

#define N_HEADS    32
#define N_KV_HEADS 8
#define HEAD_DIM   64

#if __has_builtin(__builtin_amdgcn_tensor_load_to_lds)
#define HAVE_TDM 1
#else
#define HAVE_TDM 0
#endif

// ---- WMMA helpers ---------------------------------------------------------

__device__ inline v8f wmma_f16(v16h a, v16h b, v8f c) {
  return __builtin_amdgcn_wmma_f32_16x16x32_f16(
      false, a, false, b, (short)0, c, false, false);
}

// A-matrix 16x32 f16 fragment from row-major [rows x ld].
// lane<16: row=lane, K=0..15 ; lane>=16: row=lane-16, K=16..31
__device__ inline v16h load_a_frag(const _Float16* base, int ld, int row0, int k0) {
  int lane = threadIdx.x & 31;
  int half = lane >> 4;
  int m    = lane & 15;
  const _Float16* p = base + (size_t)(row0 + m) * ld + k0 + half * 16;
  return *(const v16h*)(p);
}

// B-matrix 32x16 f16 fragment: B[k][n] = W[n0+n][k0+k], W row-major (N x ld).
// lane<16: col n=lane, K=0..15 ; lane>=16: col n=lane-16, K=16..31
__device__ inline v16h load_b_frag_wT(const _Float16* W, int ld, int n0, int k0) {
  int lane = threadIdx.x & 31;
  int n  = lane & 15;
  int kb = (lane >> 4) * 16;
  return *(const v16h*)(W + (size_t)(n0 + n) * ld + k0 + kb);
}

// ---- Tensor Data Mover: 2-D tile global -> LDS ---------------------------
// D# per CDNA5 ISA ch.8: group0 {count, lds_addr, global_addr, type=2},
// group1 {data_size=2B, tensor dims, tile dims, dim0 stride}, groups2/3 = 0.

__device__ inline void tdm_load_2d(unsigned lds_off, const void* gaddr,
                                   unsigned tile_k, unsigned tile_rows,
                                   unsigned tensor_k, unsigned tensor_rows,
                                   unsigned row_stride_elems) {
#if HAVE_TDM
  unsigned long long ga = (unsigned long long)(uintptr_t)gaddr;
  u32x4 g0;
  g0[0] = 1u;                                        // count=1, user mode
  g0[1] = lds_off;                                   // lds_addr (bytes)
  g0[2] = (unsigned)(ga & 0xffffffffu);              // global_addr[31:0]
  g0[3] = (unsigned)((ga >> 32) & 0x01ffffffu)       // global_addr[56:32]
        | (2u << 30);                                // type = 2 (image)
  i32x8 g1;
  g1[0] = (int)(1u << 16);                           // data_size=1 -> 2 bytes
  g1[1] = (int)((tensor_k & 0xffffu) << 16);         // tensor_dim0[15:0]
  g1[2] = (int)((tensor_k >> 16) & 0xffffu)          // tensor_dim0[31:16]
        | (int)((tensor_rows & 0xffffu) << 16);      // tensor_dim1[15:0]
  g1[3] = (int)((tensor_rows >> 16) & 0xffffu)       // tensor_dim1[31:16]
        | (int)((tile_k & 0xffffu) << 16);           // tile_dim0
  g1[4] = (int)(tile_rows & 0xffffu);                // tile_dim1 (tile_dim2=0)
  g1[5] = (int)row_stride_elems;                     // tensor_dim0_stride[31:0]
  g1[6] = 0;
  g1[7] = 0;
  i32x4 z4 = {0, 0, 0, 0};
#if __clang_major__ >= 23
  i32x8 z8 = {0, 0, 0, 0, 0, 0, 0, 0};
  __builtin_amdgcn_tensor_load_to_lds(g0, g1, z4, z4, z8, 0);
#else
  __builtin_amdgcn_tensor_load_to_lds(g0, g1, z4, z4, 0);
#endif
#else
  (void)lds_off; (void)gaddr; (void)tile_k; (void)tile_rows;
  (void)tensor_k; (void)tensor_rows; (void)row_stride_elems;
#endif
}

// ---- fp32 -> f16 conversion ----------------------------------------------

__global__ void convert_f32_f16(const float* __restrict__ src,
                                _Float16* __restrict__ dst, int n) {
  int i = blockIdx.x * blockDim.x + threadIdx.x;
  int stride = gridDim.x * blockDim.x;
  for (; i < n; i += stride) dst[i] = (_Float16)src[i];
}

// ---- WMMA GEMM with TDM-staged LDS tiles ----------------------------------
// C(MxN,f32) = A(MxK,f16) * W(NxK,f16)^T.  8 waves/block, 128x64 tile/block,
// wave w owns rows w*16..w*16+15; B tile shared by all 8 waves from LDS.

#define MT 128
#define NT 64
#define KT 32

__global__ __launch_bounds__(256) void gemm_f16_wmma_lds(
    const _Float16* __restrict__ A, const _Float16* __restrict__ W,
    float* __restrict__ C, int M, int N, int K) {
  __shared__ __align__(64) _Float16 ldsA[2][MT * KT];   // 2 x 8 KB
  __shared__ __align__(64) _Float16 ldsB[2][NT * KT];   // 2 x 4 KB

  int n0   = blockIdx.x * NT;
  int row0 = blockIdx.y * MT;
  int wave = threadIdx.x >> 5;

  v8f acc[4] = {};
  int iters = K / KT;

#if HAVE_TDM
  if (threadIdx.x < 32) {   // one wave issues TDM (EXEC ignored by TDM)
    tdm_load_2d((unsigned)(size_t)&ldsA[0][0], A + (size_t)row0 * K,
                KT, MT, K, M, K);
    tdm_load_2d((unsigned)(size_t)&ldsB[0][0], W + (size_t)n0 * K,
                KT, NT, K, N, K);
  }
  for (int it = 0; it < iters; ++it) {
    int buf = it & 1;
    if (threadIdx.x < 32) __builtin_amdgcn_s_wait_tensorcnt(0);
    __syncthreads();                       // buf ready; buf^1 free of readers
    if (threadIdx.x < 32 && it + 1 < iters) {
      int k0n = (it + 1) * KT;             // prefetch next tiles into buf^1
      tdm_load_2d((unsigned)(size_t)&ldsA[buf ^ 1][0],
                  A + (size_t)row0 * K + k0n, KT, MT, K, M, K);
      tdm_load_2d((unsigned)(size_t)&ldsB[buf ^ 1][0],
                  W + (size_t)n0 * K + k0n, KT, NT, K, N, K);
    }
    v16h a = load_a_frag(&ldsA[buf][0], KT, wave * 16, 0);
#pragma unroll
    for (int j = 0; j < 4; ++j)
      acc[j] = wmma_f16(a, load_b_frag_wT(&ldsB[buf][0], KT, j * 16, 0), acc[j]);
    __syncthreads();                       // compute(it) done before overwrite
  }
#else
  // Fallback: cooperative b128 copies, single buffer.
  for (int it = 0; it < iters; ++it) {
    int k0 = it * KT;
    __syncthreads();
    {
      int t = threadIdx.x;
      int row = t >> 1, hf = t & 1;        // 256 threads cover 128x32 A tile
      *(v16h*)&ldsA[0][row * KT + hf * 16] =
          *(const v16h*)(A + (size_t)(row0 + row) * K + k0 + hf * 16);
      if (t < 128) {                       // 128 threads cover 64x32 B tile
        *(v16h*)&ldsB[0][row * KT + hf * 16] =
            *(const v16h*)(W + (size_t)(n0 + row) * K + k0 + hf * 16);
      }
    }
    __syncthreads();
    v16h a = load_a_frag(&ldsA[0][0], KT, wave * 16, 0);
#pragma unroll
    for (int j = 0; j < 4; ++j)
      acc[j] = wmma_f16(a, load_b_frag_wT(&ldsB[0][0], KT, j * 16, 0), acc[j]);
  }
#endif

  int lane = threadIdx.x & 31;
  int half = lane >> 4, col = lane & 15;
  int wrow = row0 + wave * 16;
#pragma unroll
  for (int j = 0; j < 4; ++j)
#pragma unroll
    for (int r = 0; r < 8; ++r)
      C[(size_t)(wrow + r + half * 8) * N + n0 + j * 16 + col] = acc[j][r];
}

// ---- RoPE + split QKV -----------------------------------------------------
// qkv fp32 (S x 3072) -> Qh f16 (S x 2048), Kh f16 (S x 512),
// Vt f16 TRANSPOSED (512 x S): row = kv_head*64 + d, col = s.

__global__ void rope_split_kernel(const float* __restrict__ qkv,
                                  const int* __restrict__ pos,
                                  _Float16* __restrict__ Qh,
                                  _Float16* __restrict__ Kh,
                                  _Float16* __restrict__ Vt, int S) {
  int p = blockIdx.x * blockDim.x + threadIdx.x;
  int total = S * 1536;  // pairs per row: 1024 Q + 256 K + 256 V
  if (p >= total) return;
  int s = p / 1536;
  int c = p - s * 1536;
  const float* row = qkv + (size_t)s * 3072;
  float ps = (float)pos[s];
  if (c < 1024) {                       // Q pairs
    int head = c >> 5, i = c & 31;
    int col  = head * HEAD_DIM + 2 * i;
    float x1 = row[col], x2 = row[col + 1];
    float freq = __powf(10000.0f, -(2.0f * (float)i) / 64.0f);
    float sn, cs; __sincosf(ps * freq, &sn, &cs);
    Qh[(size_t)s * 2048 + col]     = (_Float16)(x1 * cs - x2 * sn);
    Qh[(size_t)s * 2048 + col + 1] = (_Float16)(x1 * sn + x2 * cs);
  } else if (c < 1280) {                // K pairs (rotated)
    int idx = c - 1024;
    int head = idx >> 5, i = idx & 31;
    int col  = head * HEAD_DIM + 2 * i;
    float x1 = row[2048 + col], x2 = row[2048 + col + 1];
    float freq = __powf(10000.0f, -(2.0f * (float)i) / 64.0f);
    float sn, cs; __sincosf(ps * freq, &sn, &cs);
    Kh[(size_t)s * 512 + col]     = (_Float16)(x1 * cs - x2 * sn);
    Kh[(size_t)s * 512 + col + 1] = (_Float16)(x1 * sn + x2 * cs);
  } else {                              // V pairs -> transposed f16
    int col = (c - 1280) * 2;
    Vt[(size_t)col * S + s]       = (_Float16)row[2560 + col];
    Vt[(size_t)(col + 1) * S + s] = (_Float16)row[2560 + col + 1];
  }
}

// ---- Flash attention: one wave per (head, 16-query tile) ------------------

__global__ __launch_bounds__(32) void flash_attn_kernel(
    const _Float16* __restrict__ Qh, const _Float16* __restrict__ Kh,
    const _Float16* __restrict__ Vt, _Float16* __restrict__ Oh, int S) {
  __shared__ __align__(32) _Float16 Ptile[16 * 32];

  int h     = blockIdx.x;               // 0..31
  int qbase = blockIdx.y * 16;
  int kvh   = h >> 2;                   // 32 heads -> 8 kv heads
  int lane  = threadIdx.x & 31;
  int half  = lane >> 4, nl = lane & 15;

  const _Float16* Qb  = Qh + h * HEAD_DIM;                 // ld 2048
  const _Float16* Kb  = Kh + kvh * HEAD_DIM;               // ld 512
  const _Float16* Vtb = Vt + (size_t)kvh * HEAD_DIM * S;   // ld S (d-major)

  v16h qa0 = load_a_frag(Qb, 2048, qbase, 0);
  v16h qa1 = load_a_frag(Qb, 2048, qbase, 32);

  v8f o[4] = {};
  float m_run[8], l_run[8];
#pragma unroll
  for (int r = 0; r < 8; ++r) { m_run[r] = -1e30f; l_run[r] = 0.0f; }

  const float scale = 0.125f;   // 1/sqrt(HEAD_DIM)
  int qmax = qbase + 15;

  for (int kb = 0; kb <= qmax; kb += 32) {
    if (kb + 32 <= qmax)      // prefetch next key tile (global_prefetch_b8)
      __builtin_prefetch(Kb + (size_t)(kb + 32) * 512, 0, 1);

    // 16q x 32k scores as two 16x16 WMMA tiles (each accumulates d=0..63)
    v8f sc0 = {}, sc1 = {};
    sc0 = wmma_f16(qa0, load_b_frag_wT(Kb, 512, kb,      0),  sc0);
    sc0 = wmma_f16(qa1, load_b_frag_wT(Kb, 512, kb,      32), sc0);
    sc1 = wmma_f16(qa0, load_b_frag_wT(Kb, 512, kb + 16, 0),  sc1);
    sc1 = wmma_f16(qa1, load_b_frag_wT(Kb, 512, kb + 16, 32), sc1);

    int key0 = kb + nl;
    int key1 = kb + 16 + nl;
#pragma unroll
    for (int r = 0; r < 8; ++r) {
      int qrow = qbase + r + half * 8;
      float v0 = sc0[r] * scale;
      float v1 = sc1[r] * scale;
      if (key0 > qrow) v0 = -1e30f;
      if (key1 > qrow) v1 = -1e30f;
      float rm = fmaxf(v0, v1);       // row reduce: xor 1..8 stays in-half
      rm = fmaxf(rm, __shfl_xor(rm, 1, 32));
      rm = fmaxf(rm, __shfl_xor(rm, 2, 32));
      rm = fmaxf(rm, __shfl_xor(rm, 4, 32));
      rm = fmaxf(rm, __shfl_xor(rm, 8, 32));
      float mnew  = fmaxf(m_run[r], rm);
      float alpha = __expf(m_run[r] - mnew);
      float p0 = __expf(v0 - mnew);
      float p1 = __expf(v1 - mnew);
      float rs = p0 + p1;
      rs += __shfl_xor(rs, 1, 32);
      rs += __shfl_xor(rs, 2, 32);
      rs += __shfl_xor(rs, 4, 32);
      rs += __shfl_xor(rs, 8, 32);
      l_run[r] = l_run[r] * alpha + rs;
      m_run[r] = mnew;
#pragma unroll
      for (int j = 0; j < 4; ++j) o[j][r] *= alpha;
      int prow = r + half * 8;
      Ptile[prow * 32 + nl]      = (_Float16)p0;
      Ptile[prow * 32 + 16 + nl] = (_Float16)p1;
    }
    __syncthreads();
    v16h pa = load_a_frag(Ptile, 32, 0, 0);      // P as A-fragment 16x32
#pragma unroll
    for (int j = 0; j < 4; ++j)                  // B[k][d] = Vt[d][key]
      o[j] = wmma_f16(pa, load_b_frag_wT(Vtb, S, j * 16, kb), o[j]);
    __syncthreads();
  }

#pragma unroll
  for (int r = 0; r < 8; ++r) {
    float inv = 1.0f / l_run[r];
    int row = qbase + r + half * 8;
#pragma unroll
    for (int j = 0; j < 4; ++j)
      Oh[(size_t)row * 2048 + h * HEAD_DIM + j * 16 + nl] =
          (_Float16)(o[j][r] * inv);
  }
}

// ---------------------------------------------------------------------------

extern "C" void kernel_launch(void* const* d_in, const int* in_sizes, int n_in,
                              void* d_out, int out_size, void* d_ws, size_t ws_size,
                              hipStream_t stream) {
  const float* x   = (const float*)d_in[0];   // (1, S, 2048)
  const int*   pos = (const int*)d_in[1];     // (S)
  const float* wq  = (const float*)d_in[2];   // (2048, 2048)
  const float* wk  = (const float*)d_in[3];   // (512, 2048)
  const float* wv  = (const float*)d_in[4];   // (512, 2048)
  const float* wo  = (const float*)d_in[5];   // (2048, 2048)
  float* out = (float*)d_out;

  const int S = in_sizes[1];
  const int D = 2048;
  const int QO = N_HEADS * HEAD_DIM;       // 2048
  const int KVO = N_KV_HEADS * HEAD_DIM;   // 512
  const int NQKV = QO + 2 * KVO;           // 3072

  char* ws = (char*)d_ws;
  size_t off = 0;
  auto carve = [&](size_t bytes) {
    char* p = ws + off;
    off += (bytes + 255) & ~(size_t)255;
    return p;
  };
  _Float16* Xh    = (_Float16*)carve((size_t)S * D * 2);
  _Float16* Wqkvh = (_Float16*)carve((size_t)NQKV * D * 2);
  _Float16* Woh   = (_Float16*)carve((size_t)D * QO * 2);
  float*    QKV   = (float*)   carve((size_t)S * NQKV * 4);
  _Float16* Qh    = (_Float16*)carve((size_t)S * QO * 2);
  _Float16* Kh    = (_Float16*)carve((size_t)S * KVO * 2);
  _Float16* Vt    = (_Float16*)carve((size_t)KVO * S * 2);   // transposed V
  _Float16* Oh    = (_Float16*)carve((size_t)S * QO * 2);
  (void)ws_size;

  const int CT = 256;
  convert_f32_f16<<<(S * D + CT - 1) / CT, CT, 0, stream>>>(x, Xh, S * D);
  convert_f32_f16<<<(QO * D + CT - 1) / CT, CT, 0, stream>>>(wq, Wqkvh, QO * D);
  convert_f32_f16<<<(KVO * D + CT - 1) / CT, CT, 0, stream>>>(
      wk, Wqkvh + (size_t)QO * D, KVO * D);
  convert_f32_f16<<<(KVO * D + CT - 1) / CT, CT, 0, stream>>>(
      wv, Wqkvh + (size_t)(QO + KVO) * D, KVO * D);
  convert_f32_f16<<<(D * QO + CT - 1) / CT, CT, 0, stream>>>(wo, Woh, D * QO);

  // QKV projection: (S x 3072) = Xh * Wqkvh^T  (TDM-staged WMMA GEMM)
  {
    dim3 grid(NQKV / NT, S / MT);
    gemm_f16_wmma_lds<<<grid, 256, 0, stream>>>(Xh, Wqkvh, QKV, S, NQKV, D);
  }
  // RoPE + split (V transposed)
  {
    int total = S * 1536;
    rope_split_kernel<<<(total + CT - 1) / CT, CT, 0, stream>>>(
        QKV, pos, Qh, Kh, Vt, S);
  }
  // flash attention
  {
    dim3 grid(N_HEADS, S / 16);
    flash_attn_kernel<<<grid, 32, 0, stream>>>(Qh, Kh, Vt, Oh, S);
  }
  // output projection: out (S x 2048) = Oh * Woh^T
  {
    dim3 grid(D / NT, S / MT);
    gemm_f16_wmma_lds<<<grid, 256, 0, stream>>>(Oh, Woh, out, S, D, QO);
  }
}
